// ScaleDotProductAttention_34866544509558
// MI455X (gfx1250) — matrix operand order repaired and checked by reading, not compiled, not run on hardware
//
#include <hip/hip_runtime.h>

typedef float  v2f  __attribute__((ext_vector_type(2)));
typedef float  v4f  __attribute__((ext_vector_type(4)));
typedef float  v8f  __attribute__((ext_vector_type(8)));
typedef int    v4i  __attribute__((ext_vector_type(4)));
typedef int    gv4i __attribute__((vector_size(16)));      // async builtin param
typedef __attribute__((ext_vector_type(16))) __bf16 v16bf;
typedef __attribute__((ext_vector_type(8)))  __bf16 v8bf;

#define B_   2
#define H_   16
#define L_   2048
#define D_   64
#define BH_  (B_ * H_)
#define KEYS_ 32                 // keys per tile (K=32 bf16 WMMA)
#define NKT_  (L_ / KEYS_)       // 64 key tiles
#define NEG_VAL (-10000.0f)
#define SCALE_  (0.125f)         // 1/sqrt(64)

#define WAVES_   4
#define THREADS_ 128
#define QROWS_   (WAVES_ * 16)   // 64 query rows per block
#define QBLKS_   (L_ / QROWS_)   // 32 q-blocks per (b,h)

#if defined(__has_builtin)
#if __has_builtin(__builtin_amdgcn_global_load_async_to_lds_b128)
#define HAVE_ASYNC_LDS 1
#endif
#if __has_builtin(__builtin_amdgcn_s_wait_asynccnt)
#define HAVE_WAIT_ASYNC 1
#endif
#endif

__device__ __forceinline__ void async_copy_b128(const void* gsrc, void* ldst)
{
#if defined(HAVE_ASYNC_LDS)
    __builtin_amdgcn_global_load_async_to_lds_b128((gv4i*)gsrc, (gv4i*)ldst, 0, 0);
#else
    *(v4f*)ldst = *(const v4f*)gsrc;
#endif
}

#if defined(HAVE_ASYNC_LDS)
#if defined(HAVE_WAIT_ASYNC)
#define WAIT_ASYNC(N) __builtin_amdgcn_s_wait_asynccnt((unsigned short)(N))
#else
#define WAIT_ASYNC(N) asm volatile("s_wait_asynccnt %0" ::"i"(N) : "memory")
#endif
#else
#define WAIT_ASYNC(N) ((void)0)
#endif

// bf16 A-fragment (16x32): elements [0..7] = K in [16hi,16hi+8),
//                          elements [8..15] = K in [16hi+8,16hi+16)
__device__ __forceinline__ v16bf make_afrag(v4f f0, v4f f1, v4f f2, v4f f3)
{
    v16bf a;
#pragma unroll
    for (int i = 0; i < 4; ++i) {
        a[i]      = (__bf16)f0[i];
        a[4 + i]  = (__bf16)f1[i];
        a[8 + i]  = (__bf16)f2[i];
        a[12 + i] = (__bf16)f3[i];
    }
    return a;
}

// ---------------------------------------------------------------------------
// Kernel 1: softmax row stats (m, l); per-lane online stats (single exp per
// element), associative 16-lane merge at the end.
// ---------------------------------------------------------------------------
__global__ __launch_bounds__(THREADS_) void attn_stats_kernel(
    const float* __restrict__ g, const float* __restrict__ e,
    const int* __restrict__ mask,
    float* __restrict__ mrow, float* __restrict__ lrow)
{
    __shared__ __align__(16) float  ebuf[2][KEYS_ * D_];   // 16KB
    __shared__ __align__(32) __bf16 ebf[KEYS_ * D_];       // 4KB
    __shared__ __align__(16) int    mbuf[QROWS_ * KEYS_];  // 8KB

    const int tid  = threadIdx.x;
    const int wave = tid >> 5;
    const int lane = tid & 31;
    const int n    = lane & 15;
    const int hi   = lane >> 4;
    const int bh   = blockIdx.x / QBLKS_;
    const int qblk = blockIdx.x % QBLKS_;
    const int q0   = qblk * QROWS_ + wave * 16;

    const float* gb = g + (size_t)bh * L_ * D_;
    const float* eb = e + (size_t)bh * L_ * D_;
    const int*   mb = mask + (size_t)(qblk * QROWS_) * L_;

    v16bf a[2];
#pragma unroll
    for (int dh = 0; dh < 2; ++dh) {
        const float* gr = gb + (size_t)(q0 + n) * D_ + dh * 32;
        a[dh] = make_afrag(*(const v4f*)(gr + 16 * hi),
                           *(const v4f*)(gr + 16 * hi + 4),
                           *(const v4f*)(gr + 16 * hi + 8),
                           *(const v4f*)(gr + 16 * hi + 12));
    }

#pragma unroll
    for (int j = 0; j < 4; ++j) {
        const int idx = tid + THREADS_ * j;
        async_copy_b128(eb + idx * 4, &ebuf[0][idx * 4]);
    }

    float m_ln[8], l_ln[8];
#pragma unroll
    for (int r = 0; r < 8; ++r) { m_ln[r] = -3.0e38f; l_ln[r] = 0.0f; }

    for (int kt = 0; kt < NKT_; ++kt) {
        const int buf = kt & 1;
        const int ktn = (kt + 1 < NKT_) ? kt + 1 : kt;
        const int kb  = kt * KEYS_;
#pragma unroll
        for (int j = 0; j < 4; ++j) {
            const int idx = tid + THREADS_ * j;
            async_copy_b128(eb + (size_t)ktn * KEYS_ * D_ + idx * 4,
                            &ebuf[buf ^ 1][idx * 4]);
        }
        WAIT_ASYNC(4);
        __syncthreads();    // B1: f32 tile kt ready

        // ---- convert: e f32 -> bf16 LDS; mask global -> LDS ----
#pragma unroll
        for (int j = 0; j < 2; ++j) {
            const int bidx = (tid + THREADS_ * j) * 8;
            v4f f0 = *(const v4f*)&ebuf[buf][bidx];
            v4f f1 = *(const v4f*)&ebuf[buf][bidx + 4];
            v8bf h;
#pragma unroll
            for (int i = 0; i < 4; ++i) { h[i] = (__bf16)f0[i]; h[4 + i] = (__bf16)f1[i]; }
            *(v8bf*)&ebf[bidx] = h;
        }
#pragma unroll
        for (int j = 0; j < 4; ++j) {
            const int idx = tid + THREADS_ * j;
            const int rr = idx >> 3, qd = idx & 7;
            *(v4i*)&mbuf[rr * KEYS_ + qd * 4] =
                *(const v4i*)(mb + (size_t)rr * L_ + kb + qd * 4);
        }
        __syncthreads();    // B2

        // ---- S tile: dh outer / kg inner -> 2 independent WMMA chains,
        //      adjacent instructions share A => reuse_a hint legal ----
        v8f c[2] = {};
#pragma unroll
        for (int dh = 0; dh < 2; ++dh) {
            v16bf b0 = *(const v16bf*)&ebf[(0 * 16 + n) * D_ + dh * 32 + 16 * hi];
            v16bf b1 = *(const v16bf*)&ebf[(1 * 16 + n) * D_ + dh * 32 + 16 * hi];
            c[0] = __builtin_amdgcn_wmma_f32_16x16x32_bf16(
                       false, a[dh], false, b0, (short)0, c[0], false, false);
            c[1] = __builtin_amdgcn_wmma_f32_16x16x32_bf16(
                       false, a[dh], false, b1, (short)0, c[1], true, false);
        }
        // ---- online per-lane softmax stats: one exp per element ----
#pragma unroll
        for (int kg = 0; kg < 2; ++kg)
#pragma unroll
            for (int r = 0; r < 8; ++r) {
                float sv = c[kg][r] * SCALE_;
                if (mbuf[(wave * 16 + hi * 8 + r) * KEYS_ + kg * 16 + n] == 0) sv = NEG_VAL;
                const float m_old = m_ln[r];
                const float m_new = fmaxf(m_old, sv);
                const float t = __expf(fminf(m_old, sv) - m_new);  // exp(0)=1 case folded
                const bool  svbig = sv >= m_old;
                l_ln[r] = l_ln[r] * (svbig ? t : 1.0f) + (svbig ? 1.0f : t);
                m_ln[r] = m_new;
            }
    }

    // associative (m,l) merge across the 16 lanes of each half
#pragma unroll
    for (int r = 0; r < 8; ++r) {
        float m = m_ln[r], l = l_ln[r];
#pragma unroll
        for (int k = 1; k <= 8; k <<= 1) {
            const float mo = __shfl_xor(m, k);
            const float lo = __shfl_xor(l, k);
            const float mm = fmaxf(m, mo);
            l = l * __expf(m - mm) + lo * __expf(mo - mm);
            m = mm;
        }
        if (n == 0) {
            const int row = q0 + hi * 8 + r;
            mrow[(size_t)bh * L_ + row] = m;
            lrow[(size_t)bh * L_ + row] = l;
        }
    }
}

// ---------------------------------------------------------------------------
// Kernel 2: recompute S (bf16 WMMA), normalize, NT b128 score stores,
// O = P @ x with xT bf16 tiles; PV WMMAs share A (reuse hints).
// ---------------------------------------------------------------------------
__global__ __launch_bounds__(THREADS_) void attn_out_kernel(
    const float* __restrict__ g, const float* __restrict__ e,
    const float* __restrict__ x, const int* __restrict__ mask,
    const float* __restrict__ mrow, const float* __restrict__ lrow,
    float* __restrict__ out, float* __restrict__ score)
{
    __shared__ __align__(16) float  ebuf[2][KEYS_ * D_];    // 16KB
    __shared__ __align__(16) float  xbuf[2][KEYS_ * D_];    // 16KB
    __shared__ __align__(32) __bf16 ebf[KEYS_ * D_];        // 4KB
    __shared__ __align__(32) __bf16 xtbuf[D_ * KEYS_];      // 4KB  [d][key]
    __shared__ __align__(16) int    mbuf[QROWS_ * KEYS_];   // 8KB
    __shared__ __align__(16) float  pbuf[WAVES_][16 * 36];  // 9KB

    const int tid  = threadIdx.x;
    const int wave = tid >> 5;
    const int lane = tid & 31;
    const int n    = lane & 15;
    const int hi   = lane >> 4;
    const int bh   = blockIdx.x / QBLKS_;
    const int qblk = blockIdx.x % QBLKS_;
    const int q0   = qblk * QROWS_ + wave * 16;

    const float* gb = g + (size_t)bh * L_ * D_;
    const float* eb = e + (size_t)bh * L_ * D_;
    const float* xb = x + (size_t)bh * L_ * D_;
    const int*   mb = mask + (size_t)(qblk * QROWS_) * L_;
    float* wp = &pbuf[wave][0];

    v16bf a[2];
#pragma unroll
    for (int dh = 0; dh < 2; ++dh) {
        const float* gr = gb + (size_t)(q0 + n) * D_ + dh * 32;
        a[dh] = make_afrag(*(const v4f*)(gr + 16 * hi),
                           *(const v4f*)(gr + 16 * hi + 4),
                           *(const v4f*)(gr + 16 * hi + 8),
                           *(const v4f*)(gr + 16 * hi + 12));
    }

    float mr[8], li[8];
#pragma unroll
    for (int r = 0; r < 8; ++r) {
        const int row = q0 + hi * 8 + r;
        mr[r] = mrow[(size_t)bh * L_ + row];
        li[r] = 1.0f / lrow[(size_t)bh * L_ + row];
    }

    v8f o[4];
#pragma unroll
    for (int c4 = 0; c4 < 4; ++c4) o[c4] = (v8f){};

#pragma unroll
    for (int j = 0; j < 4; ++j) {
        const int idx = tid + THREADS_ * j;
        async_copy_b128(eb + idx * 4, &ebuf[0][idx * 4]);
        async_copy_b128(xb + idx * 4, &xbuf[0][idx * 4]);
    }

    for (int kt = 0; kt < NKT_; ++kt) {
        const int buf = kt & 1;
        const int ktn = (kt + 1 < NKT_) ? kt + 1 : kt;
        const int kb  = kt * KEYS_;
#pragma unroll
        for (int j = 0; j < 4; ++j) {
            const int idx = tid + THREADS_ * j;
            async_copy_b128(eb + (size_t)ktn * KEYS_ * D_ + idx * 4, &ebuf[buf ^ 1][idx * 4]);
            async_copy_b128(xb + (size_t)ktn * KEYS_ * D_ + idx * 4, &xbuf[buf ^ 1][idx * 4]);
        }
        WAIT_ASYNC(8);
        __syncthreads();    // B1

        // ---- convert: e -> bf16; x -> bf16 transposed; mask -> LDS ----
#pragma unroll
        for (int j = 0; j < 2; ++j) {
            const int bidx = (tid + THREADS_ * j) * 8;
            v4f f0 = *(const v4f*)&ebuf[buf][bidx];
            v4f f1 = *(const v4f*)&ebuf[buf][bidx + 4];
            v8bf h;
#pragma unroll
            for (int i = 0; i < 4; ++i) { h[i] = (__bf16)f0[i]; h[4 + i] = (__bf16)f1[i]; }
            *(v8bf*)&ebf[bidx] = h;
        }
#pragma unroll
        for (int j = 0; j < 2; ++j) {
            const int s = tid + THREADS_ * j;      // 0..255
            const int d = s & 63, kg8 = s >> 6;
            v8bf h;
#pragma unroll
            for (int i = 0; i < 8; ++i)
                h[i] = (__bf16)xbuf[buf][(kg8 * 8 + i) * D_ + d];
            *(v8bf*)&xtbuf[d * KEYS_ + kg8 * 8] = h;
        }
#pragma unroll
        for (int j = 0; j < 4; ++j) {
            const int idx = tid + THREADS_ * j;
            const int rr = idx >> 3, qd = idx & 7;
            *(v4i*)&mbuf[rr * KEYS_ + qd * 4] =
                *(const v4i*)(mb + (size_t)rr * L_ + kb + qd * 4);
        }
        __syncthreads();    // B2

        // ---- S tile: 2 independent chains, reuse_a on 2nd of each pair ----
        v8f c[2] = {};
#pragma unroll
        for (int dh = 0; dh < 2; ++dh) {
            v16bf b0 = *(const v16bf*)&ebf[(0 * 16 + n) * D_ + dh * 32 + 16 * hi];
            v16bf b1 = *(const v16bf*)&ebf[(1 * 16 + n) * D_ + dh * 32 + 16 * hi];
            c[0] = __builtin_amdgcn_wmma_f32_16x16x32_bf16(
                       false, a[dh], false, b0, (short)0, c[0], false, false);
            c[1] = __builtin_amdgcn_wmma_f32_16x16x32_bf16(
                       false, a[dh], false, b1, (short)0, c[1], true, false);
        }
        // ---- normalize, stage P ----
#pragma unroll
        for (int kg = 0; kg < 2; ++kg)
#pragma unroll
            for (int r = 0; r < 8; ++r) {
                float sv = c[kg][r] * SCALE_;
                if (mbuf[(wave * 16 + hi * 8 + r) * KEYS_ + kg * 16 + n] == 0) sv = NEG_VAL;
                wp[(hi * 8 + r) * 36 + kg * 16 + n] = __expf(sv - mr[r]) * li[r];
            }
        // ---- score: 4 x non-temporal b128 stores per lane ----
#pragma unroll
        for (int i = 0; i < 4; ++i) {
            const int slot = lane + 32 * i;
            const int rl = slot >> 3, qs = slot & 7;
            v4f pv = *(const v4f*)&wp[rl * 36 + qs * 4];
            __builtin_nontemporal_store(
                pv, (v4f*)&score[(size_t)(bh * L_ + q0 + rl) * L_ + kb + qs * 4]);
        }
        // ---- O += P @ x: 4 independent chains sharing A; peel c4=0, then
        //      reuse_a=true literal for c4=1..3 (builtin needs ICE) ----
        v16bf pa = make_afrag(*(const v4f*)&wp[n * 36 + 16 * hi],
                              *(const v4f*)&wp[n * 36 + 16 * hi + 4],
                              *(const v4f*)&wp[n * 36 + 16 * hi + 8],
                              *(const v4f*)&wp[n * 36 + 16 * hi + 12]);
        {
            v16bf bxf = *(const v16bf*)&xtbuf[(0 * 16 + n) * KEYS_ + 16 * hi];
            o[0] = __builtin_amdgcn_wmma_f32_16x16x32_bf16(
                       false, pa, false, bxf, (short)0, o[0], false, false);
        }
#pragma unroll
        for (int c4 = 1; c4 < 4; ++c4) {
            v16bf bxf = *(const v16bf*)&xtbuf[(c4 * 16 + n) * KEYS_ + 16 * hi];
            o[c4] = __builtin_amdgcn_wmma_f32_16x16x32_bf16(
                        false, pa, false, bxf, (short)0, o[c4], true, false);
        }
    }

    // ---- write out tile [16 x 64] (non-temporal) ----
#pragma unroll
    for (int c4 = 0; c4 < 4; ++c4)
#pragma unroll
        for (int r = 0; r < 8; ++r) {
            const int row = q0 + hi * 8 + r;
            __builtin_nontemporal_store(
                o[c4][r], &out[(size_t)(bh * L_ + row) * D_ + c4 * 16 + n]);
        }
}

// ---------------------------------------------------------------------------
extern "C" void kernel_launch(void* const* d_in, const int* in_sizes, int n_in,
                              void* d_out, int out_size, void* d_ws, size_t ws_size,
                              hipStream_t stream)
{
    (void)in_sizes; (void)n_in; (void)out_size; (void)ws_size;
    const float* g    = (const float*)d_in[0];
    const float* e    = (const float*)d_in[1];
    const float* x    = (const float*)d_in[2];
    const int*   mask = (const int*)d_in[3];

    float* out   = (float*)d_out;
    float* score = out + (size_t)B_ * H_ * L_ * D_;

    float* mrow = (float*)d_ws;
    float* lrow = mrow + (size_t)BH_ * L_;

    dim3 grid(BH_ * QBLKS_);   // 1024 blocks
    dim3 blk(THREADS_);        // 4 waves; one 16-row q-tile per wave

    hipLaunchKernelGGL(attn_stats_kernel, grid, blk, 0, stream,
                       g, e, mask, mrow, lrow);
    hipLaunchKernelGGL(attn_out_kernel, grid, blk, 0, stream,
                       g, e, x, mask, mrow, lrow, out, score);
}
